// MultiheadSelfAttentionV1_40724879900973
// MI455X (gfx1250) — compile-verified, matrix operand order checked by reading, not run on hardware
//
#include <hip/hip_runtime.h>

// MI455X / gfx1250 multi-head self-attention forward.
// B=8, T=1024, F=1024, H=16, D=64.  out = [B,T,F] f32, alpha = [B,H,T,T] f32.
// All four GEMMs on v_wmma_f32_16x16x32_bf16; attention core fully fused so
// the T*T matrix touches HBM exactly once (the mandatory alpha output).

typedef __attribute__((ext_vector_type(16))) __bf16 v16bf;
typedef __attribute__((ext_vector_type(8)))  __bf16 v8bf;
typedef __attribute__((ext_vector_type(8)))  float  v8f;

// ISA 7.12.2: 16-bit A-matrix 16x32 element mapping for lane half `hi`:
//  element e -> K = (e/8)*16 + hi*8 + (e%8)   (two contiguous 8-elem chunks)
#define KIDX(e, hi) ((((e) >> 3) << 4) + ((hi) << 3) + ((e) & 7))

__device__ __forceinline__ v8f wmma_bf16(v16bf a, v16bf b, v8f c) {
  return __builtin_amdgcn_wmma_f32_16x16x32_bf16(false, a, false, b,
                                                 (short)0, c, false, false);
}

__device__ __forceinline__ v8f vzero8() {
  v8f z = {0.f, 0.f, 0.f, 0.f, 0.f, 0.f, 0.f, 0.f};
  return z;
}

// Vectorized fragment load: one 32-wide K slice = two 16-byte chunks at
// byte offsets 16*hi and 32+16*hi.  Requires p to be 16B aligned.
__device__ __forceinline__ v16bf load_frag16(const __bf16* p, int hi) {
  const v8bf c0 = *(const v8bf*)(p + (hi << 3));
  const v8bf c1 = *(const v8bf*)(p + 16 + (hi << 3));
  return __builtin_shufflevector(c0, c1, 0, 1, 2, 3, 4, 5, 6, 7,
                                 8, 9, 10, 11, 12, 13, 14, 15);
}

// ---------------------------------------------------------------------------
// Kernel 1: fused QKV projection.
// C[8192,3072] = X[8192,1024] @ W[3072,1024]^T + bias, scattered to
//   Q  [B,H,T,64] bf16, K [B,H,T,64] bf16, Vt [B,H,64,T] bf16 (transposed).
// Block 256 thr = 8 waves; block tile 128x64; wave tile 32x32; K step 64.
// ---------------------------------------------------------------------------
__global__ void __launch_bounds__(256)
k_qkv_proj(const float* __restrict__ x,    // [8192,1024]
           const float* __restrict__ w,    // [3072,1024]
           const float* __restrict__ bias, // [3072]
           __bf16* __restrict__ Qb, __bf16* __restrict__ Kb,
           __bf16* __restrict__ Vt) {
  const int AL = 66;  // padded LDS stride (132B = 33 banks, gcd(33,64)=1)
  __shared__ __bf16 At[128 * 66];
  __shared__ __bf16 Bt[64 * 66];

  const int tid = threadIdx.x;
  const int w32 = tid >> 5, lane = tid & 31, l16 = lane & 15, hi = lane >> 4;
  const int wm = w32 & 3, wn = w32 >> 2;
  const int mBase = blockIdx.x * 128;
  const int nBase = blockIdx.y * 64;

  v8f c[2][2];
  c[0][0] = vzero8(); c[0][1] = vzero8(); c[1][0] = vzero8(); c[1][1] = vzero8();

  const int r0 = tid >> 4;         // 0..15
  const int c4 = (tid & 15) << 2;  // 0..60 step 4

  for (int k0 = 0; k0 < 1024; k0 += 64) {
    // stage A tile 128x64 (fp32 -> bf16)
#pragma unroll
    for (int rr = 0; rr < 8; ++rr) {
      const int r = r0 + rr * 16;
      const float4 v = *(const float4*)(x + (size_t)(mBase + r) * 1024 + k0 + c4);
      __bf16* dst = &At[r * AL + c4];
      dst[0] = (__bf16)v.x; dst[1] = (__bf16)v.y;
      dst[2] = (__bf16)v.z; dst[3] = (__bf16)v.w;
    }
    // stage B tile 64x64 (rows of W; x@W^T means B[k][n] = W[n][k])
#pragma unroll
    for (int rr = 0; rr < 4; ++rr) {
      const int r = r0 + rr * 16;
      const float4 v = *(const float4*)(w + (size_t)(nBase + r) * 1024 + k0 + c4);
      __bf16* dst = &Bt[r * AL + c4];
      dst[0] = (__bf16)v.x; dst[1] = (__bf16)v.y;
      dst[2] = (__bf16)v.z; dst[3] = (__bf16)v.w;
    }
    if (k0 + 64 < 1024) {  // global_prefetch_b8 for next A tile
      __builtin_prefetch(x + (size_t)(mBase + r0) * 1024 + k0 + 64 + c4, 0, 1);
    }
    __syncthreads();

    v16bf a[2][2], b[2][2];
#pragma unroll
    for (int ms = 0; ms < 2; ++ms) {
      const __bf16* p = &At[(wm * 32 + ms * 16 + l16) * AL];
#pragma unroll
      for (int kh = 0; kh < 2; ++kh)
#pragma unroll
        for (int e = 0; e < 16; ++e) a[ms][kh][e] = p[kh * 32 + KIDX(e, hi)];
    }
#pragma unroll
    for (int ns = 0; ns < 2; ++ns) {
      const __bf16* p = &Bt[(wn * 32 + ns * 16 + l16) * AL];
#pragma unroll
      for (int kh = 0; kh < 2; ++kh)
#pragma unroll
        for (int e = 0; e < 16; ++e) b[ns][kh][e] = p[kh * 32 + KIDX(e, hi)];
    }
#pragma unroll
    for (int kh = 0; kh < 2; ++kh)
#pragma unroll
      for (int ms = 0; ms < 2; ++ms)
#pragma unroll
        for (int ns = 0; ns < 2; ++ns)
          c[ms][ns] = wmma_bf16(a[ms][kh], b[ns][kh], c[ms][ns]);
    __syncthreads();
  }

  // epilogue: bias + scatter to Q / K / Vt
#pragma unroll
  for (int ns = 0; ns < 2; ++ns) {
    const int n = nBase + wn * 32 + ns * 16 + l16;  // 0..3071
    const float bn = bias[n];
    const int region = n >> 10;  // 0=Q 1=K 2=V
    const int nn = n & 1023;
    const int h = nn >> 6, d = nn & 63;
#pragma unroll
    for (int ms = 0; ms < 2; ++ms) {
#pragma unroll
      for (int r = 0; r < 8; ++r) {
        const int m = mBase + wm * 32 + ms * 16 + hi * 8 + r;
        const int bb = m >> 10, t = m & 1023;
        const __bf16 val = (__bf16)(c[ms][ns][r] + bn);
        if (region == 0)
          Qb[((size_t)(bb * 16 + h) * 1024 + t) * 64 + d] = val;
        else if (region == 1)
          Kb[((size_t)(bb * 16 + h) * 1024 + t) * 64 + d] = val;
        else
          Vt[((size_t)(bb * 16 + h) * 64 + d) * 1024 + t] = val;
      }
    }
  }
}

// ---------------------------------------------------------------------------
// Kernel 2: fused attention core for one (b,h) head and one 16-row query tile.
// 8 waves each own a 128-key slice of the 1024-wide score row:
//   scores (QK^T, 16 WMMA) -> scale+mask -> 2-level softmax (shfl + LDS)
//   -> alpha to d_out (f32) + LDS (bf16) -> alphaV (16 WMMA) -> LDS reduce.
// Grid: (T/16=64, B*H=128).
// ---------------------------------------------------------------------------
__global__ void __launch_bounds__(256)
k_attention(const __bf16* __restrict__ Qb, const __bf16* __restrict__ Kb,
            const __bf16* __restrict__ Vt, const unsigned char* __restrict__ mask,
            float* __restrict__ alpha, __bf16* __restrict__ Att) {
  const int AL = 1032;  // padded alpha stride (2064B = 129*16B: keeps 16B align)
  __shared__ __align__(16) unsigned char smem[16 * 1032 * 2 + 2 * 128 * 4];
  __bf16* aLds = (__bf16*)smem;                      // 16 x 1032 bf16 (33024 B)
  float* redmax = (float*)(smem + 16 * 1032 * 2);    // [8][16]
  float* redsum = redmax + 128;                      // [8][16]
  float* avp = (float*)smem;                         // overlap: [8][16][64] f32

  const int tid = threadIdx.x;
  const int w32 = tid >> 5, lane = tid & 31, l16 = lane & 15, hi = lane >> 4;
  const int qt = blockIdx.x;  // query tile
  const int bh = blockIdx.y;  // b*16 + h
  const int b = bh >> 4;
  const size_t base = (size_t)bh << 16;  // head stride 1024*64 = 65536
  const int cw = w32 * 128;              // this wave's key-column base

  // Q fragments (rows qt*16..+15, d 0..63); rows are 128B aligned
  const __bf16* qp = Qb + base + (size_t)(qt * 16 + l16) * 64;
  const v16bf aq0 = load_frag16(qp, hi);
  const v16bf aq1 = load_frag16(qp + 32, hi);

  // ---- scores: S = Q K^T over this wave's 128 key columns (8 n-tiles) ----
  v8f c[8];
#pragma unroll
  for (int nt = 0; nt < 8; ++nt) {
    v8f acc = vzero8();
    const __bf16* kp = Kb + base + (size_t)(cw + nt * 16 + l16) * 64;
    acc = wmma_bf16(aq0, load_frag16(kp, hi), acc);
    acc = wmma_bf16(aq1, load_frag16(kp + 32, hi), acc);
    c[nt] = acc;
  }

  // ---- scale (1/sqrt(64)) + key-padding mask (column-wise) ----
  const unsigned char* mrow = mask + b * 1024;
#pragma unroll
  for (int nt = 0; nt < 8; ++nt) {
    const bool msk = mrow[cw + nt * 16 + l16] != 0;
#pragma unroll
    for (int r = 0; r < 8; ++r) {
      const float v = c[nt][r] * 0.125f;
      c[nt][r] = msk ? -1.0e30f : v;
    }
  }

  // ---- row max: lane-local, shfl over 16-lane half, LDS over waves ----
  float rm[8];
#pragma unroll
  for (int r = 0; r < 8; ++r) {
    float m = c[0][r];
#pragma unroll
    for (int nt = 1; nt < 8; ++nt) m = fmaxf(m, c[nt][r]);
#pragma unroll
    for (int off = 1; off < 16; off <<= 1) m = fmaxf(m, __shfl_xor(m, off, 32));
    rm[r] = m;
  }
  if (l16 == 0) {
#pragma unroll
    for (int r = 0; r < 8; ++r) redmax[w32 * 16 + hi * 8 + r] = rm[r];
  }
  __syncthreads();
  float gm[8];
#pragma unroll
  for (int r = 0; r < 8; ++r) {
    float m = -3.0e38f;
#pragma unroll
    for (int ww = 0; ww < 8; ++ww) m = fmaxf(m, redmax[ww * 16 + hi * 8 + r]);
    gm[r] = m;
  }

  // ---- exp + row sum (same two-level reduction) ----
  float rs[8];
#pragma unroll
  for (int r = 0; r < 8; ++r) rs[r] = 0.f;
#pragma unroll
  for (int nt = 0; nt < 8; ++nt) {
#pragma unroll
    for (int r = 0; r < 8; ++r) {
      const float e = __expf(c[nt][r] - gm[r]);
      c[nt][r] = e;
      rs[r] += e;
    }
  }
#pragma unroll
  for (int r = 0; r < 8; ++r) {
    float s = rs[r];
#pragma unroll
    for (int off = 1; off < 16; off <<= 1) s += __shfl_xor(s, off, 32);
    rs[r] = s;
  }
  if (l16 == 0) {
#pragma unroll
    for (int r = 0; r < 8; ++r) redsum[w32 * 16 + hi * 8 + r] = rs[r];
  }
  __syncthreads();
  float inv[8];
#pragma unroll
  for (int r = 0; r < 8; ++r) {
    float s = 0.f;
#pragma unroll
    for (int ww = 0; ww < 8; ++ww) s += redsum[ww * 16 + hi * 8 + r];
    inv[r] = 1.0f / s;
  }

  // ---- normalize; write alpha to d_out (f32) and to LDS (bf16) ----
  float* arow = alpha + ((size_t)bh * 1024 + qt * 16) * 1024;
#pragma unroll
  for (int nt = 0; nt < 8; ++nt) {
    const int col = cw + nt * 16 + l16;
#pragma unroll
    for (int r = 0; r < 8; ++r) {
      const int row = hi * 8 + r;
      const float a = c[nt][r] * inv[r];
      arow[(size_t)row * 1024 + col] = a;
      aLds[row * AL + col] = (__bf16)a;
    }
  }

  // ---- alphaV: partial [16 x 64] over this wave's 128 keys (4 k-steps) ----
  v8f c2[4];
  c2[0] = vzero8(); c2[1] = vzero8(); c2[2] = vzero8(); c2[3] = vzero8();
#pragma unroll
  for (int ks = 0; ks < 4; ++ks) {
    const int kofs = cw + ks * 32;
    const v16bf af = load_frag16(&aLds[l16 * AL + kofs], hi);  // 16B aligned
#pragma unroll
    for (int nt = 0; nt < 4; ++nt) {
      const __bf16* vp = Vt + base + (size_t)(nt * 16 + l16) * 1024 + kofs;
      c2[nt] = wmma_bf16(af, load_frag16(vp, hi), c2[nt]);
    }
  }

  __syncthreads();  // everyone done reading aLds; safe to overlap with avp
#pragma unroll
  for (int nt = 0; nt < 4; ++nt) {
#pragma unroll
    for (int r = 0; r < 8; ++r)
      avp[(w32 * 16 + hi * 8 + r) * 64 + nt * 16 + l16] = c2[nt][r];
  }
  __syncthreads();

  // ---- cross-wave reduce + store att tile into [B,T,F] workspace (bf16) ----
  const int h = bh & 15;
#pragma unroll
  for (int j = 0; j < 4; ++j) {
    const int idx = tid + j * 256;  // 0..1023 over 16x64 tile
    const int row = idx >> 6, d = idx & 63;
    float s = 0.f;
#pragma unroll
    for (int ww = 0; ww < 8; ++ww) s += avp[(ww * 16 + row) * 64 + d];
    Att[((size_t)(b * 1024 + qt * 16 + row)) * 1024 + h * 64 + d] = (__bf16)s;
  }
}

// ---------------------------------------------------------------------------
// Kernel 3: output projection.  out[8192,1024] = Att @ Wout^T + bias (f32 out)
// Same tiling as K1 (K step 64); A is already bf16 in workspace.
// ---------------------------------------------------------------------------
__global__ void __launch_bounds__(256)
k_out_proj(const __bf16* __restrict__ A,   // [8192,1024] bf16
           const float* __restrict__ w,    // [1024,1024]
           const float* __restrict__ bias, // [1024]
           float* __restrict__ out) {      // [8192,1024]
  const int AL = 66;
  __shared__ __bf16 At[128 * 66];
  __shared__ __bf16 Bt[64 * 66];

  const int tid = threadIdx.x;
  const int w32 = tid >> 5, lane = tid & 31, l16 = lane & 15, hi = lane >> 4;
  const int wm = w32 & 3, wn = w32 >> 2;
  const int mBase = blockIdx.x * 128;
  const int nBase = blockIdx.y * 64;

  v8f c[2][2];
  c[0][0] = vzero8(); c[0][1] = vzero8(); c[1][0] = vzero8(); c[1][1] = vzero8();

  const int ra = tid >> 3;         // 0..31 (A rows per pass)
  const int ca = (tid & 7) << 3;   // 0..56 step 8
  const int rb = tid >> 4;         // 0..15
  const int cb = (tid & 15) << 2;  // 0..60 step 4

  for (int k0 = 0; k0 < 1024; k0 += 64) {
#pragma unroll
    for (int rr = 0; rr < 4; ++rr) {
      const int r = ra + rr * 32;
      const __bf16* src = A + (size_t)(mBase + r) * 1024 + k0 + ca;
      __bf16* dst = &At[r * AL + ca];
#pragma unroll
      for (int q = 0; q < 8; ++q) dst[q] = src[q];
    }
#pragma unroll
    for (int rr = 0; rr < 4; ++rr) {
      const int r = rb + rr * 16;
      const float4 v = *(const float4*)(w + (size_t)(nBase + r) * 1024 + k0 + cb);
      __bf16* dst = &Bt[r * AL + cb];
      dst[0] = (__bf16)v.x; dst[1] = (__bf16)v.y;
      dst[2] = (__bf16)v.z; dst[3] = (__bf16)v.w;
    }
    if (k0 + 64 < 1024) {
      __builtin_prefetch(A + (size_t)(mBase + ra) * 1024 + k0 + 64 + ca, 0, 1);
    }
    __syncthreads();

    v16bf a[2][2], b[2][2];
#pragma unroll
    for (int ms = 0; ms < 2; ++ms) {
      const __bf16* p = &At[(wm * 32 + ms * 16 + l16) * AL];
#pragma unroll
      for (int kh = 0; kh < 2; ++kh)
#pragma unroll
        for (int e = 0; e < 16; ++e) a[ms][kh][e] = p[kh * 32 + KIDX(e, hi)];
    }
#pragma unroll
    for (int ns = 0; ns < 2; ++ns) {
      const __bf16* p = &Bt[(wn * 32 + ns * 16 + l16) * AL];
#pragma unroll
      for (int kh = 0; kh < 2; ++kh)
#pragma unroll
        for (int e = 0; e < 16; ++e) b[ns][kh][e] = p[kh * 32 + KIDX(e, hi)];
    }
#pragma unroll
    for (int kh = 0; kh < 2; ++kh)
#pragma unroll
      for (int ms = 0; ms < 2; ++ms)
#pragma unroll
        for (int ns = 0; ns < 2; ++ns)
          c[ms][ns] = wmma_bf16(a[ms][kh], b[ns][kh], c[ms][ns]);
    __syncthreads();
  }

#pragma unroll
  for (int ns = 0; ns < 2; ++ns) {
    const int n = nBase + wn * 32 + ns * 16 + l16;
    const float bn = bias[n];
#pragma unroll
    for (int ms = 0; ms < 2; ++ms) {
#pragma unroll
      for (int r = 0; r < 8; ++r) {
        const int m = mBase + wm * 32 + ms * 16 + hi * 8 + r;
        out[(size_t)m * 1024 + n] = c[ms][ns][r] + bn;
      }
    }
  }
}

// ---------------------------------------------------------------------------
extern "C" void kernel_launch(void* const* d_in, const int* in_sizes, int n_in,
                              void* d_out, int out_size, void* d_ws, size_t ws_size,
                              hipStream_t stream) {
  const float* qkv = (const float*)d_in[0];                  // [8,1024,1024] f32
  const unsigned char* mask = (const unsigned char*)d_in[1]; // [8,1024] bool
  const float* ipw = (const float*)d_in[2];                  // [3072,1024] f32
  const float* ipb = (const float*)d_in[3];                  // [3072] f32
  const float* opw = (const float*)d_in[4];                  // [1024,1024] f32
  const float* opb = (const float*)d_in[5];                  // [1024] f32

  float* out = (float*)d_out;                    // [8,1024,1024]
  float* alpha = out + (size_t)8 * 1024 * 1024;  // [8,16,1024,1024]

  // Workspace: 4 bf16 buffers of B*T*F = 8,388,608 elems each (64 MB total).
  __bf16* Qb = (__bf16*)d_ws;                  // [B,H,T,64]
  __bf16* Kb = Qb + (size_t)8 * 1024 * 1024;   // [B,H,T,64]
  __bf16* Vt = Kb + (size_t)8 * 1024 * 1024;   // [B,H,64,T] (transposed)
  __bf16* Att = Vt + (size_t)8 * 1024 * 1024;  // [B,T,F]

  k_qkv_proj<<<dim3(64, 48), 256, 0, stream>>>(qkv, ipw, ipb, Qb, Kb, Vt);
  k_attention<<<dim3(64, 128), 256, 0, stream>>>(Qb, Kb, Vt, mask, alpha, Att);
  k_out_proj<<<dim3(64, 16), 256, 0, stream>>>(Att, opw, opb, out);
}
